// Spatial_MOSI_triple_64836826300484
// MI455X (gfx1250) — matrix operand order repaired and checked by reading, not compiled, not run on hardware
//
#include <hip/hip_runtime.h>
#include <hip/hip_bf16.h>

typedef __bf16 v16bf __attribute__((ext_vector_type(16)));
typedef float  v8f   __attribute__((ext_vector_type(8)));

#define NN   20000
#define EE   320000
#define HH   512
#define DD   30

__device__ __forceinline__ float eluf(float x) { return x > 0.f ? x : (expf(x) - 1.f); }

// ---------------------------------------------------------------------------
// WMMA GEMM: C[M,N] = op(A[M,K]) @ B[K,N] (+bias) (elu)
// flags: bit0 = apply ELU to A elements, bit1 = add bias[N], bit2 = ELU on output
// Block: 256 threads (8 waves). Tile 64x128, K-step 32 (bf16 WMMA).
// Double-buffered LDS: stage tile t+1 while WMMAs consume tile t (1 barrier/iter).
// Each wave: one 16x32 A fragment reused across four 32x16 B fragments -> 4 WMMA.
// ---------------------------------------------------------------------------
#define GTM 64
#define GTN 128
#define GTK 32

template <bool FULLK>
__device__ __forceinline__ void stage_tiles(
    const float* __restrict__ A, const float* __restrict__ B,
    int M, int N, int K, int k0, int tileM, int tileN, bool aelu,
    __bf16 (&lA)[GTM][GTK], __bf16 (&lB)[GTN][GTK + 8], int tid)
{
    // ---- A tile (64 x 32): 256 thr x float4 x 2 iters ----
    {
        int kq = (tid & 7) * 4;
        int mr = tid >> 3;                       // 0..31
        #pragma unroll
        for (int it = 0; it < 2; ++it, mr += 32) {
            int gm = tileM + mr;
            float4 f = make_float4(0.f, 0.f, 0.f, 0.f);
            if (FULLK) {
                if (gm < M)
                    f = *(const float4*)(A + (size_t)gm * K + (k0 + kq));
            } else {
                if (gm < M) {
                    const float* s = A + (size_t)gm * K + (k0 + kq);
                    int gk = k0 + kq;
                    if (gk + 0 < K) f.x = s[0];
                    if (gk + 1 < K) f.y = s[1];
                    if (gk + 2 < K) f.z = s[2];
                    if (gk + 3 < K) f.w = s[3];
                }
            }
            if (aelu) { f.x = eluf(f.x); f.y = eluf(f.y); f.z = eluf(f.z); f.w = eluf(f.w); }
            lA[mr][kq + 0] = (__bf16)f.x;
            lA[mr][kq + 1] = (__bf16)f.y;
            lA[mr][kq + 2] = (__bf16)f.z;
            lA[mr][kq + 3] = (__bf16)f.w;
        }
    }
    // ---- B tile (32 x 128) transposed into lB[n][k]: 256 thr x float4 x 4 iters ----
    {
        int nq  = (tid & 31) * 4;
        int kr0 = tid >> 5;                      // 0..7
        bool nvec = (N & 3) == 0;                // float4 path alignment-safe
        #pragma unroll
        for (int it = 0; it < 4; ++it) {
            int kr = kr0 + it * 8;
            int gk = k0 + kr;
            int gn = tileN + nq;
            float4 f = make_float4(0.f, 0.f, 0.f, 0.f);
            if (FULLK) {                         // gk < K guaranteed
                if (nvec && (gn + 4 <= N)) {
                    f = *(const float4*)(B + (size_t)gk * N + gn);
                } else {
                    const float* s = B + (size_t)gk * N + gn;
                    if (gn + 0 < N) f.x = s[0];
                    if (gn + 1 < N) f.y = s[1];
                    if (gn + 2 < N) f.z = s[2];
                    if (gn + 3 < N) f.w = s[3];
                }
            } else {
                if (gk < K) {
                    const float* s = B + (size_t)gk * N + gn;
                    if (gn + 0 < N) f.x = s[0];
                    if (gn + 1 < N) f.y = s[1];
                    if (gn + 2 < N) f.z = s[2];
                    if (gn + 3 < N) f.w = s[3];
                }
            }
            lB[nq + 0][kr] = (__bf16)f.x;
            lB[nq + 1][kr] = (__bf16)f.y;
            lB[nq + 2][kr] = (__bf16)f.z;
            lB[nq + 3][kr] = (__bf16)f.w;
        }
    }
}

__global__ __launch_bounds__(256) void wmma_gemm_kernel(
    const float* __restrict__ A, const float* __restrict__ B,
    const float* __restrict__ bias, float* __restrict__ C,
    int M, int N, int K, int flags)
{
    __shared__ __bf16 lA[2][GTM][GTK];       //  8 KB
    __shared__ __bf16 lB[2][GTN][GTK + 8];   // 20 KB (transposed + padded)

    const int tid  = threadIdx.x;
    const int lane = tid & 31;
    const int wave = tid >> 5;
    const int wm   = wave & 3;   // 0..3 -> M sub-tile (x16)
    const int wn   = wave >> 2;  // 0..1 -> N half (x64)
    const int tileM = blockIdx.y * GTM;
    const int tileN = blockIdx.x * GTN;
    const bool aelu = (flags & 1) != 0;

    // full 32-wide K tiles eligible for the vectorized path
    const int kfull = ((K & 3) == 0) ? (K & ~(GTK - 1)) : 0;

    v8f acc[4] = {};

    // ---- prologue: stage tile 0 into buffer 0 ----
    if (0 < kfull) stage_tiles<true >(A, B, M, N, K, 0, tileM, tileN, aelu, lA[0], lB[0], tid);
    else           stage_tiles<false>(A, B, M, N, K, 0, tileM, tileN, aelu, lA[0], lB[0], tid);
    __syncthreads();

    int buf = 0;
    for (int k0 = 0; k0 < K; k0 += GTK) {
        // ---- stage next tile into the alternate buffer (overlaps with WMMA) ----
        int k1 = k0 + GTK;
        if (k1 < K) {
            if (k1 < kfull)
                stage_tiles<true >(A, B, M, N, K, k1, tileM, tileN, aelu,
                                   lA[buf ^ 1], lB[buf ^ 1], tid);
            else
                stage_tiles<false>(A, B, M, N, K, k1, tileM, tileN, aelu,
                                   lA[buf ^ 1], lB[buf ^ 1], tid);
        }

        // ---- consume current buffer ----
        // A fragment: 16x32 bf16 per ISA layout
        int ar = wm * 16 + (lane & 15);
        int kb = (lane >> 4) * 8;
        union { uint4 u[2]; v16bf v; } af;
        af.u[0] = *(const uint4*)&lA[buf][ar][kb];
        af.u[1] = *(const uint4*)&lA[buf][ar][kb + 16];

        // 4 B fragments (32x16 each), one WMMA per fragment
        int bkk   = (lane >> 4) * 8;
        int cbase = wn * 64 + (lane & 15);
        #pragma unroll
        for (int sub = 0; sub < 4; ++sub) {
            int c = cbase + sub * 16;
            union { uint4 u[2]; v16bf v; } bf;
            bf.u[0] = *(const uint4*)&lB[buf][c][bkk];
            bf.u[1] = *(const uint4*)&lB[buf][c][bkk + 16];
            acc[sub] = __builtin_amdgcn_wmma_f32_16x16x32_bf16(
                false, af.v, false, bf.v, (short)0, acc[sub], false, false);
        }

        // single barrier: next buffer fully staged AND current buffer consumed
        __syncthreads();
        buf ^= 1;
    }

    // ---- epilogue: C/D layout row = v + (lane>=16)*8, col = lane&15 ----
    int rbase = tileM + wm * 16 + ((lane >> 4) * 8);
    #pragma unroll
    for (int sub = 0; sub < 4; ++sub) {
        int col = tileN + wn * 64 + sub * 16 + (lane & 15);
        if (col >= N) continue;
        float b = (flags & 2) ? bias[col] : 0.f;
        #pragma unroll
        for (int v = 0; v < 8; ++v) {
            int r = rbase + v;
            if (r >= M) continue;
            float x = acc[sub][v] + b;
            if (flags & 4) x = eluf(x);
            C[(size_t)r * N + col] = x;
        }
    }
}

// ---------------------------------------------------------------------------
// Elementwise fill (grid-stride)
// ---------------------------------------------------------------------------
__global__ void fill_kernel(float* __restrict__ p, float v, long long n)
{
    long long i = (long long)blockIdx.x * blockDim.x + threadIdx.x;
    long long st = (long long)gridDim.x * blockDim.x;
    for (; i < n; i += st) p[i] = v;
}

// ---------------------------------------------------------------------------
// Per-row dual dot products: als[r] = h[r]·a_s, ald[r] = h[r]·a_d (wave/row)
// ---------------------------------------------------------------------------
__global__ __launch_bounds__(256) void rowdot2_kernel(
    const float* __restrict__ h, const float* __restrict__ a_s,
    const float* __restrict__ a_d, float* __restrict__ als,
    float* __restrict__ ald, int n)
{
    int wave = threadIdx.x >> 5, lane = threadIdx.x & 31;
    int row = blockIdx.x * 8 + wave;
    if (row >= n) return;
    const float* hp = h + (size_t)row * HH;
    float s0 = 0.f, s1 = 0.f;
    for (int i = lane; i < HH; i += 32) {
        float v = hp[i];
        s0 = fmaf(v, a_s[i], s0);
        s1 = fmaf(v, a_d[i], s1);
    }
    for (int off = 16; off; off >>= 1) {
        s0 += __shfl_down(s0, off, 32);
        s1 += __shfl_down(s1, off, 32);
    }
    if (lane == 0) { als[row] = s0; ald[row] = s1; }
}

__device__ __forceinline__ float leaky02(float x) { return x > 0.f ? x : 0.2f * x; }

__device__ __forceinline__ void atomicMaxF(float* addr, float v)
{
    if (v >= 0.f) atomicMax((int*)addr, __float_as_int(v));
    else          atomicMin((unsigned int*)addr, __float_as_uint(v));
}

// ---------------------------------------------------------------------------
// Edge softmax passes
// ---------------------------------------------------------------------------
__global__ void edge_max_kernel(const int* __restrict__ ei,
                                const float* __restrict__ als,
                                const float* __restrict__ ald,
                                float* __restrict__ mx)
{
    int e = blockIdx.x * blockDim.x + threadIdx.x;
    if (e >= EE) return;
    int s = ei[e], d = ei[EE + e];
    atomicMaxF(&mx[d], leaky02(als[s] + ald[d]));
}

__global__ void edge_expsum_kernel(const int* __restrict__ ei,
                                   const float* __restrict__ als,
                                   const float* __restrict__ ald,
                                   const float* __restrict__ mx,
                                   float* __restrict__ eexp,
                                   float* __restrict__ ssum)
{
    int e = blockIdx.x * blockDim.x + threadIdx.x;
    if (e >= EE) return;
    int s = ei[e], d = ei[EE + e];
    float v = expf(leaky02(als[s] + ald[d]) - mx[d]);
    eexp[e] = v;
    atomicAdd(&ssum[d], v);
}

// one block (128 thr) per edge: agg[d] += alpha * h[s]   (H=512, float4/thread)
__global__ __launch_bounds__(128) void edge_aggregate_kernel(
    const int* __restrict__ ei, const float* __restrict__ eexp,
    const float* __restrict__ ssum, const float* __restrict__ h,
    float* __restrict__ agg)
{
    int e = blockIdx.x;
    int s = ei[e], d = ei[EE + e];
    float alpha = eexp[e] / (ssum[d] + 1e-16f);
    const float4 v = ((const float4*)(h + (size_t)s * HH))[threadIdx.x];
    float* op = agg + (size_t)d * HH + threadIdx.x * 4;
    atomicAdd(op + 0, v.x * alpha);
    atomicAdd(op + 1, v.y * alpha);
    atomicAdd(op + 2, v.z * alpha);
    atomicAdd(op + 3, v.w * alpha);
}

// ---------------------------------------------------------------------------
// Cross-modal attention fusion (wave per node, lanes 0..29 hold D dims)
// ---------------------------------------------------------------------------
__global__ __launch_bounds__(128) void fusion_kernel(
    const float* __restrict__ h2a, const float* __restrict__ h2b,
    const float* __restrict__ h2c, const float* __restrict__ w_omega,
    const float* __restrict__ u_omega, float* __restrict__ emb, int n)
{
    int wave = threadIdx.x >> 5, lane = threadIdx.x & 31;
    int node = blockIdx.x * 4 + wave;
    if (node >= n) return;
    float e0 = (lane < DD) ? h2a[(size_t)node * DD + lane] : 0.f;
    float e1 = (lane < DD) ? h2b[(size_t)node * DD + lane] : 0.f;
    float e2 = (lane < DD) ? h2c[(size_t)node * DD + lane] : 0.f;
    float v0 = 0.f, v1 = 0.f, v2 = 0.f;
    for (int d = 0; d < DD; ++d) {
        float w = (lane < DD) ? w_omega[d * DD + lane] : 0.f;
        v0 = fmaf(__shfl(e0, d, 32), w, v0);
        v1 = fmaf(__shfl(e1, d, 32), w, v1);
        v2 = fmaf(__shfl(e2, d, 32), w, v2);
    }
    v0 = tanhf(v0); v1 = tanhf(v1); v2 = tanhf(v2);
    float u = (lane < DD) ? u_omega[lane] : 0.f;
    float s0 = v0 * u, s1 = v1 * u, s2 = v2 * u;
    for (int off = 16; off; off >>= 1) {
        s0 += __shfl_xor(s0, off, 32);
        s1 += __shfl_xor(s1, off, 32);
        s2 += __shfl_xor(s2, off, 32);
    }
    s0 += 1e-6f; s1 += 1e-6f; s2 += 1e-6f;
    float mmax = fmaxf(s0, fmaxf(s1, s2));
    float a0 = expf(s0 - mmax), a1 = expf(s1 - mmax), a2 = expf(s2 - mmax);
    float inv = 1.f / (a0 + a1 + a2);
    if (lane < DD)
        emb[(size_t)node * DD + lane] = (a0 * e0 + a1 * e1 + a2 * e2) * inv;
}

// ---------------------------------------------------------------------------
// CSL scatter-mean: psum[ei0] += h2[ei1]; cnt[ei0]++; then divide
// ---------------------------------------------------------------------------
__global__ __launch_bounds__(256) void csl_accum_kernel(
    const int* __restrict__ ei, const float* __restrict__ h2,
    float* __restrict__ psum, float* __restrict__ cnt)
{
    int wave = threadIdx.x >> 5, lane = threadIdx.x & 31;
    int e = blockIdx.x * 8 + wave;
    if (e >= EE) return;
    int a = ei[e];         // ei[0] : segment id
    int b = ei[EE + e];    // ei[1] : source row
    if (lane < DD) atomicAdd(&psum[(size_t)a * DD + lane], h2[(size_t)b * DD + lane]);
    if (lane == 0) atomicAdd(&cnt[a], 1.f);
}

__global__ void csl_div_kernel(float* __restrict__ p, const float* __restrict__ cnt, int n)
{
    int i = blockIdx.x * blockDim.x + threadIdx.x;
    if (i >= n * DD) return;
    p[i] /= fmaxf(cnt[i / DD], 1.f);
}

// ---------------------------------------------------------------------------
extern "C" void kernel_launch(void* const* d_in, const int* in_sizes, int n_in,
                              void* d_out, int out_size, void* d_ws, size_t ws_size,
                              hipStream_t stream)
{
    (void)in_sizes; (void)n_in; (void)out_size; (void)ws_size;
    const int N = NN, E = EE, H = HH, D = DD;
    const int INs[3] = {3000, 1000, 500};

    const float* feat[3] = {(const float*)d_in[0], (const float*)d_in[1], (const float*)d_in[2]};
    const int*   ei[3]   = {(const int*)d_in[3], (const int*)d_in[4], (const int*)d_in[5]};
    const int*   eCSL    = (const int*)d_in[6];
    const float* W1[3]   = {(const float*)d_in[7],  (const float*)d_in[11], (const float*)d_in[15]};
    const float* a_s[3]  = {(const float*)d_in[8],  (const float*)d_in[12], (const float*)d_in[16]};
    const float* a_d[3]  = {(const float*)d_in[9],  (const float*)d_in[13], (const float*)d_in[17]};
    const float* W2[3]   = {(const float*)d_in[10], (const float*)d_in[14], (const float*)d_in[18]};
    const float* w_omega = (const float*)d_in[19];
    const float* u_omega = (const float*)d_in[20];
    const float* d1w[3]  = {(const float*)d_in[21], (const float*)d_in[25], (const float*)d_in[29]};
    const float* d1b[3]  = {(const float*)d_in[22], (const float*)d_in[26], (const float*)d_in[30]};
    const float* d2w[3]  = {(const float*)d_in[23], (const float*)d_in[27], (const float*)d_in[31]};
    const float* d2b[3]  = {(const float*)d_in[24], (const float*)d_in[28], (const float*)d_in[32]};

    // workspace layout (floats)
    float* ws = (float*)d_ws;
    size_t o = 0;
    float* hbuf[3];
    for (int m = 0; m < 3; ++m) { hbuf[m] = ws + o; o += (size_t)N * H; }
    float* agg  = ws + o; o += (size_t)N * H;   // also decoder hidden
    float* als  = ws + o; o += N;
    float* ald  = ws + o; o += N;
    float* mx   = ws + o; o += N;
    float* ssum = ws + o; o += N;
    float* cnt  = ws + o; o += N;
    float* eexp = ws + o; o += E;

    // output layout (floats, return order)
    float* out = (float*)d_out;
    float* h2o[3]  = {out, out + (size_t)N * D, out + (size_t)2 * N * D};
    float* po[3]   = {out + (size_t)3 * N * D, out + (size_t)4 * N * D, out + (size_t)5 * N * D};
    float* h2no[3] = {out + (size_t)6 * N * D, out + (size_t)7 * N * D, out + (size_t)8 * N * D};
    float* embo    = out + (size_t)9 * N * D;
    float* reco[3];
    reco[0] = out + (size_t)10 * N * D;
    reco[1] = reco[0] + (size_t)N * INs[0];
    reco[2] = reco[1] + (size_t)N * INs[1];

    auto gemm = [&](const float* A, const float* B, const float* bias, float* C,
                    int M, int Nn, int K, int flags) {
        dim3 grid((Nn + GTN - 1) / GTN, (M + GTM - 1) / GTM);
        wmma_gemm_kernel<<<grid, 256, 0, stream>>>(A, B, bias, C, M, Nn, K, flags);
    };
    auto fill = [&](float* p, float v, long long n) {
        fill_kernel<<<2048, 256, 0, stream>>>(p, v, n);
    };

    const float NEG_INF = -3.402823466e38f;

    // ---- per modality: h = X@W1 (once); attention+aggregation for positive
    //      (edge_index_m) and negative (edge_CSL) graphs; h2 = elu(agg)@W2 ----
    for (int m = 0; m < 3; ++m) {
        gemm(feat[m], W1[m], nullptr, hbuf[m], N, H, INs[m], 0);
        rowdot2_kernel<<<(N + 7) / 8, 256, 0, stream>>>(hbuf[m], a_s[m], a_d[m], als, ald, N);

        for (int pass = 0; pass < 2; ++pass) {
            const int* eidx = (pass == 0) ? ei[m] : eCSL;
            float* target   = (pass == 0) ? h2o[m] : h2no[m];
            fill(mx, NEG_INF, N);
            fill(ssum, 0.f, N);
            fill(agg, 0.f, (long long)N * H);
            edge_max_kernel<<<(E + 255) / 256, 256, 0, stream>>>(eidx, als, ald, mx);
            edge_expsum_kernel<<<(E + 255) / 256, 256, 0, stream>>>(eidx, als, ald, mx, eexp, ssum);
            edge_aggregate_kernel<<<E, 128, 0, stream>>>(eidx, eexp, ssum, hbuf[m], agg);
            gemm(agg, W2[m], nullptr, target, N, D, H, /*A_ELU*/1);
        }
    }

    // ---- CSL positive aggregation: p_m = scatter_mean(h2_m[ei1], ei0) ----
    for (int m = 0; m < 3; ++m) {
        fill(po[m], 0.f, (long long)N * D);
        fill(cnt, 0.f, N);
        csl_accum_kernel<<<(E + 7) / 8, 256, 0, stream>>>(ei[m], h2o[m], po[m], cnt);
        csl_div_kernel<<<(N * D + 255) / 256, 256, 0, stream>>>(po[m], cnt, N);
    }

    // ---- cross-modal attention fusion ----
    fusion_kernel<<<(N + 3) / 4, 128, 0, stream>>>(h2o[0], h2o[1], h2o[2],
                                                   w_omega, u_omega, embo, N);

    // ---- decoders: t = elu(emb@d1w + b1); rec = t@d2w + b2 ----
    for (int m = 0; m < 3; ++m) {
        gemm(embo, d1w[m], d1b[m], agg, N, H, D, /*bias|eluOut*/ 2 | 4);
        gemm(agg, d2w[m], d2b[m], reco[m], N, INs[m], H, /*bias*/ 2);
    }
}
